// EnhancedHeteroGNN_13572096655643
// MI455X (gfx1250) — matrix operand order repaired and checked by reading, not compiled, not run on hardware
//
#include <hip/hip_runtime.h>
#include <hip/hip_fp16.h>

// ---------------------------------------------------------------------------
// EnhancedHeteroGNN for MI455X (gfx1250, wave32).
// - Dense GEMMs: v_wmma_f32_16x16x32_f16 (f16 in, f32 accum), LDS-transposed
//   coalesced b128 stores.
// - GAT softmax aggregation: CSR-by-destination gather (register accumulation,
//   no float atomics). CSR built once per edge type with an LDS block scan.
// ---------------------------------------------------------------------------

typedef __attribute__((ext_vector_type(16))) _Float16     v16h;
typedef __attribute__((ext_vector_type(8)))  float        v8f;
typedef __attribute__((ext_vector_type(4)))  unsigned int u32x4;

union Frag16 { u32x4 q[2]; v16h v; };          // 16 halves = 8 VGPRs
union H8     { u32x4 q;    _Float16 h[8]; };   // 8 halves  = 16 bytes

#define SLOPE 0.2f
#define SCAN_B 1024

__device__ __forceinline__ float lrelu(float x){ return x > 0.f ? x : SLOPE * x; }

// ------------------------------- utilities ---------------------------------

__global__ void k_fill_i32(int* p, int v, int n){
  int t = blockIdx.x * blockDim.x + threadIdx.x;
  if (t < n) p[t] = v;
}
__global__ void k_zero_f32(float* p, int n){
  int t = blockIdx.x * blockDim.x + threadIdx.x;
  if (t < n) p[t] = 0.f;
}
__global__ void k_ew(const float* ea, float* ew){
  if (blockIdx.x == 0 && threadIdx.x == 0){
    float m  = fmaxf(ea[0], fmaxf(ea[1], ea[2]));
    float e0 = __expf(ea[0]-m), e1 = __expf(ea[1]-m), e2 = __expf(ea[2]-m);
    float s  = e0 + e1 + e2;
    ew[0] = e0/s; ew[1] = e1/s; ew[2] = e2/s;
  }
}

// x = [node_features | emb[node_type]] padded to 32 cols, f16, row-major
__global__ void k_build_x(const float* __restrict__ nf, const int* __restrict__ nt,
                          const float* __restrict__ emb, _Float16* __restrict__ xh, int N){
  int t = blockIdx.x * blockDim.x + threadIdx.x;
  if (t >= N * 32) return;
  int n = t >> 5, c = t & 31;
  float v = 0.f;
  if (c < 7)       v = nf[n * 7 + c];
  else if (c < 23) v = emb[nt[n] * 16 + (c - 7)];
  xh[t] = (_Float16)v;
}

// Pack [K,64] f32 weight (zero-padded to mult of 32) into per-lane WMMA B frags:
// Bp[((ks*4+nt)*32 + lane)*16 + e], k = ks*32 + (lane>>4)*16 + e, n = nt*16+(lane&15)
__global__ void k_pack_w(const float* __restrict__ W, _Float16* __restrict__ Bp, int K){
  int ksteps = (K + 31) / 32;
  int total  = ksteps * 4 * 32 * 16;
  int t = blockIdx.x * blockDim.x + threadIdx.x;
  if (t >= total) return;
  int e = t & 15, lane = (t >> 4) & 31, ntl = (t >> 9) & 3, ks = t >> 11;
  int k = ks * 32 + (lane >> 4) * 16 + e;
  int n = ntl * 16 + (lane & 15);
  Bp[t] = (_Float16)((k < K) ? W[k * 64 + n] : 0.f);
}

// ------------------------------- WMMA GEMM ----------------------------------
// Out[N,64] = A[N,lda] x W. 4 waves/block, 16 rows/wave, LDS-transposed stores.
template<int KSTEPS>
__global__ void k_gemm(const _Float16* __restrict__ A, int lda,
                       const _Float16* __restrict__ Bp,
                       _Float16* __restrict__ Out, int nrows){
  __shared__ _Float16 sh[4][16][64];
  int lane = threadIdx.x & 31;
  int wave = threadIdx.x >> 5;
  int m0   = blockIdx.x * 64 + wave * 16;
  bool active = (m0 < nrows);              // nrows % 16 == 0 here
  int hl = lane >> 4;                      // lane half -> K sub-block / row block
  int mr = lane & 15;                      // A row in tile / D column

  if (active){
    const _Float16* arow = A + (long)(m0 + mr) * lda;
    v8f acc0 = {}, acc1 = {}, acc2 = {}, acc3 = {};
    #pragma unroll
    for (int ks = 0; ks < KSTEPS; ++ks){
      // A 16x32 f16 fragment: two contiguous b128 loads per lane.
      Frag16 af;
      const u32x4* ap = (const u32x4*)(arow + ks * 32 + hl * 16);
      af.q[0] = ap[0];
      af.q[1] = ap[1];
      const u32x4* bbase = (const u32x4*)Bp + ks * 256 + lane * 2; // nt stride = 64 u32x4
      Frag16 bf;
      bf.q[0] = bbase[0];   bf.q[1] = bbase[1];
      acc0 = __builtin_amdgcn_wmma_f32_16x16x32_f16(false, af.v, false, bf.v, (short)0, acc0, false, false);
      bf.q[0] = bbase[64];  bf.q[1] = bbase[65];
      acc1 = __builtin_amdgcn_wmma_f32_16x16x32_f16(false, af.v, false, bf.v, (short)0, acc1, false, false);
      bf.q[0] = bbase[128]; bf.q[1] = bbase[129];
      acc2 = __builtin_amdgcn_wmma_f32_16x16x32_f16(false, af.v, false, bf.v, (short)0, acc2, false, false);
      bf.q[0] = bbase[192]; bf.q[1] = bbase[193];
      acc3 = __builtin_amdgcn_wmma_f32_16x16x32_f16(false, af.v, false, bf.v, (short)0, acc3, false, false);
    }
    // D layout: lane holds column mr, VGPR r holds row r + hl*8 -> LDS transpose.
    #pragma unroll
    for (int r = 0; r < 8; ++r){
      sh[wave][hl * 8 + r][mr +  0] = (_Float16)acc0[r];
      sh[wave][hl * 8 + r][mr + 16] = (_Float16)acc1[r];
      sh[wave][hl * 8 + r][mr + 32] = (_Float16)acc2[r];
      sh[wave][hl * 8 + r][mr + 48] = (_Float16)acc3[r];
    }
  }
  __syncthreads();
  if (active){
    int row = lane >> 1, h64 = lane & 1;   // each lane stores a 64B half-row
    const u32x4* sp = (const u32x4*)&sh[wave][row][h64 * 32];
    u32x4* gp = (u32x4*)(Out + (long)(m0 + row) * 64 + h64 * 32);
    gp[0] = sp[0]; gp[1] = sp[1]; gp[2] = sp[2]; gp[3] = sp[3];
  }
}

// ---------------------------- attention logits ------------------------------

__global__ void k_al4(const _Float16* __restrict__ hx, const float* __restrict__ as,
                      const float* __restrict__ ad, float* __restrict__ al_s,
                      float* __restrict__ al_d, int N){
  int t = blockIdx.x * blockDim.x + threadIdx.x;
  if (t >= N * 4) return;
  int n = t >> 2, h = t & 3;
  const _Float16* row = hx + (long)n * 64 + h * 16;
  float s = 0.f, d = 0.f;
  #pragma unroll
  for (int c = 0; c < 16; ++c){ float v = (float)row[c]; s += v * as[h*16+c]; d += v * ad[h*16+c]; }
  al_s[t] = s; al_d[t] = d;
}
__global__ void k_al1(const _Float16* __restrict__ hx, const float* __restrict__ as,
                      const float* __restrict__ ad, float* __restrict__ al_s,
                      float* __restrict__ al_d, int N){
  int n = blockIdx.x * blockDim.x + threadIdx.x;
  if (n >= N) return;
  const _Float16* row = hx + (long)n * 64;
  float s = 0.f, d = 0.f;
  #pragma unroll
  for (int c = 0; c < 64; ++c){ float v = (float)row[c]; s += v * as[c]; d += v * ad[c]; }
  al_s[n] = s; al_d[n] = d;
}

// --------------------------- CSR-by-destination -----------------------------
// Edge id space [0, E+N): ids >= E are self-loops (src = dst = id - E).

__global__ void k_count(const int* __restrict__ ei, int E, int N, int* __restrict__ cnt){
  int t = blockIdx.x * blockDim.x + threadIdx.x;
  if (t >= E + N) return;
  int d = (t < E) ? ei[E + t] : (t - E);
  atomicAdd(&cnt[d], 1);
}

// Inclusive scan per 1024-block (Hillis-Steele in LDS) + block totals.
__global__ void k_scan_blk(const int* __restrict__ in, int* __restrict__ out,
                           int* __restrict__ bsums, int n){
  __shared__ int s[SCAN_B];
  int tid = threadIdx.x;
  int g   = blockIdx.x * SCAN_B + tid;
  int v   = (g < n) ? in[g] : 0;
  s[tid] = v;
  for (int off = 1; off < SCAN_B; off <<= 1){
    __syncthreads();
    int x = (tid >= off) ? s[tid - off] : 0;
    __syncthreads();
    s[tid] += x;
  }
  if (g < n) out[g] = s[tid];
  if (tid == SCAN_B - 1) bsums[blockIdx.x] = s[tid];
}
// Exclusive scan of block totals (nb <= 1024).
__global__ void k_scan_top(int* bsums, int nb){
  __shared__ int s[SCAN_B];
  int tid = threadIdx.x;
  int v = (tid < nb) ? bsums[tid] : 0;
  s[tid] = v;
  for (int off = 1; off < SCAN_B; off <<= 1){
    __syncthreads();
    int x = (tid >= off) ? s[tid - off] : 0;
    __syncthreads();
    s[tid] += x;
  }
  if (tid < nb) bsums[tid] = s[tid] - v;
}
__global__ void k_scan_add(int* out, const int* __restrict__ bsums, int n){
  int g = blockIdx.x * SCAN_B + threadIdx.x;
  if (g < n) out[g] += bsums[blockIdx.x];
}

__global__ void k_fillcsr(const int* __restrict__ ei, int E, int N,
                          const int* __restrict__ offs, const int* __restrict__ cnt,
                          int* __restrict__ cur, int* __restrict__ list){
  int t = blockIdx.x * blockDim.x + threadIdx.x;
  if (t >= E + N) return;
  int s, d;
  if (t < E){ s = ei[t]; d = ei[E + t]; } else { s = d = t - E; }
  int pos = atomicAdd(&cur[d], 1);
  list[offs[d] - cnt[d] + pos] = s;           // store source node id
}

// ----------------- fused softmax-aggregate (gather, no atomics) -------------
// One thread per (dst, 16-channel slice). Layer 1 (H=4, head==slice):
// out = elu(sum_src alpha*h[src] + b1) -> f16.
__global__ void k_agg1(const int* __restrict__ offs, const int* __restrict__ cnt,
                       const int* __restrict__ list,
                       const float* __restrict__ al_s, const float* __restrict__ al_d,
                       const _Float16* __restrict__ hx, const float* __restrict__ b1,
                       _Float16* __restrict__ out, int N){
  int t = blockIdx.x * blockDim.x + threadIdx.x;
  if (t >= N * 4) return;
  int n = t >> 2, part = t & 3;
  int deg   = cnt[n];
  int start = offs[n] - deg;
  float ad  = al_d[n * 4 + part];
  float mx = -1e30f;
  for (int k = 0; k < deg; ++k){
    int s = list[start + k];
    mx = fmaxf(mx, lrelu(al_s[s * 4 + part] + ad));
  }
  float den = 0.f;
  float acc[16];
  #pragma unroll
  for (int c = 0; c < 16; ++c) acc[c] = 0.f;
  for (int k = 0; k < deg; ++k){
    int s = list[start + k];
    float ex = __expf(lrelu(al_s[s * 4 + part] + ad) - mx);
    den += ex;
    const _Float16* hs = hx + (long)s * 64 + part * 16;
    H8 a, b;
    a.q = *(const u32x4*)hs;
    b.q = *(const u32x4*)(hs + 8);
    #pragma unroll
    for (int c = 0; c < 8; ++c) acc[c]     += (float)a.h[c] * ex;
    #pragma unroll
    for (int c = 0; c < 8; ++c) acc[8 + c] += (float)b.h[c] * ex;
  }
  float inv = 1.f / den;
  H8 w0, w1;
  #pragma unroll
  for (int c = 0; c < 16; ++c){
    float v = acc[c] * inv + b1[part * 16 + c];
    v = v > 0.f ? v : (__expf(v) - 1.f);               // ELU
    if (c < 8) w0.h[c] = (_Float16)v; else w1.h[c - 8] = (_Float16)v;
  }
  u32x4* op = (u32x4*)(out + (long)n * 64 + part * 16); // coalesced 32B/lane
  op[0] = w0.q; op[1] = w1.q;
}

// Layer 2 (H=1): x_comb += (sum alpha*h[src] + b2) * ew[i]
__global__ void k_agg2(const int* __restrict__ offs, const int* __restrict__ cnt,
                       const int* __restrict__ list,
                       const float* __restrict__ al_s, const float* __restrict__ al_d,
                       const _Float16* __restrict__ hx, const float* __restrict__ b2,
                       const float* __restrict__ ew, int i,
                       float* __restrict__ xcomb, int N){
  int t = blockIdx.x * blockDim.x + threadIdx.x;
  if (t >= N * 4) return;
  int n = t >> 2, part = t & 3;
  int deg   = cnt[n];
  int start = offs[n] - deg;
  float ad  = al_d[n];
  float mx = -1e30f;
  for (int k = 0; k < deg; ++k){
    int s = list[start + k];
    mx = fmaxf(mx, lrelu(al_s[s] + ad));
  }
  float den = 0.f;
  float acc[16];
  #pragma unroll
  for (int c = 0; c < 16; ++c) acc[c] = 0.f;
  for (int k = 0; k < deg; ++k){
    int s = list[start + k];
    float ex = __expf(lrelu(al_s[s] + ad) - mx);
    den += ex;
    const _Float16* hs = hx + (long)s * 64 + part * 16;
    H8 a, b;
    a.q = *(const u32x4*)hs;
    b.q = *(const u32x4*)(hs + 8);
    #pragma unroll
    for (int c = 0; c < 8; ++c) acc[c]     += (float)a.h[c] * ex;
    #pragma unroll
    for (int c = 0; c < 8; ++c) acc[8 + c] += (float)b.h[c] * ex;
  }
  float inv = 1.f / den;
  float w   = ew[i];
  float* xc = xcomb + (long)n * 64 + part * 16;
  #pragma unroll
  for (int c = 0; c < 16; ++c)
    xc[c] += (acc[c] * inv + b2[part * 16 + c]) * w;
}

// ------------------------------- readout ------------------------------------

__global__ void k_qsum(const int* __restrict__ ei, int E, int R,
                       const float* __restrict__ xcomb,
                       float* __restrict__ sums, float* __restrict__ cnts){
  int t = blockIdx.x * blockDim.x + threadIdx.x;
  if (t >= E) return;
  int s = ei[t], d = ei[E + t];
  if (s < R && d >= R){
    atomicAdd(&cnts[s], 1.f);
    const float* row = xcomb + (long)d * 64;
    float* o = sums + (long)s * 64;
    #pragma unroll
    for (int c = 0; c < 64; ++c) atomicAdd(&o[c], row[c]);
  }
}

__global__ void k_mlp1(const float* __restrict__ xcomb, const float* __restrict__ sums,
                       const float* __restrict__ cnts, const float* __restrict__ Wo1,
                       const float* __restrict__ bo1, float* __restrict__ hq1, int R){
  int t = blockIdx.x * blockDim.x + threadIdx.x;
  if (t >= R * 64) return;
  int r = t >> 6, j = t & 63;
  float s = bo1[j];
  const float* xr = xcomb + (long)r * 64;
  #pragma unroll 8
  for (int k = 0; k < 64; ++k) s += xr[k] * Wo1[k * 64 + j];
  float inv = 1.f / fmaxf(cnts[r], 1.f);
  const float* sr = sums + (long)r * 64;
  #pragma unroll 8
  for (int k = 0; k < 64; ++k) s += (sr[k] * inv) * Wo1[(64 + k) * 64 + j];
  hq1[t] = fmaxf(s, 0.f);
}
__global__ void k_mlp2(const float* __restrict__ hq1, const float* __restrict__ Wo2,
                       const float* __restrict__ bo2, float* __restrict__ hq2, int R){
  int t = blockIdx.x * blockDim.x + threadIdx.x;
  if (t >= R * 32) return;
  int r = t >> 5, j = t & 31;
  float s = bo2[j];
  const float* hr = hq1 + (long)r * 64;
  #pragma unroll 8
  for (int k = 0; k < 64; ++k) s += hr[k] * Wo2[k * 32 + j];
  hq2[t] = fmaxf(s, 0.f);
}
__global__ void k_mlp3(const float* __restrict__ hq2, const float* __restrict__ Wo3,
                       const float* __restrict__ bo3, float* __restrict__ out, int R){
  int t = blockIdx.x * blockDim.x + threadIdx.x;
  if (t >= R * 10) return;
  int r = t / 10, j = t % 10;
  float s = bo3[j];
  const float* hr = hq2 + (long)r * 32;
  #pragma unroll
  for (int k = 0; k < 32; ++k) s += hr[k] * Wo3[k * 10 + j];
  out[t] = s;
}

// ------------------------------- launcher -----------------------------------

static inline unsigned cdiv(long a, long b){ return (unsigned)((a + b - 1) / b); }

extern "C" void kernel_launch(void* const* d_in, const int* in_sizes, int n_in,
                              void* d_out, int out_size, void* d_ws, size_t ws_size,
                              hipStream_t stream){
  const float* nf    = (const float*)d_in[0];
  const int*   ntyp  = (const int*)  d_in[1];
  const int*   ei3[3]= { (const int*)d_in[2], (const int*)d_in[3], (const int*)d_in[4] };
  const float* emb   = (const float*)d_in[5];
  const float* eattn = (const float*)d_in[6];
  const float* W1    = (const float*)d_in[7];
  const float* as1   = (const float*)d_in[8];
  const float* ad1   = (const float*)d_in[9];
  const float* b1    = (const float*)d_in[10];
  const float* W2    = (const float*)d_in[11];
  const float* as2   = (const float*)d_in[12];
  const float* ad2   = (const float*)d_in[13];
  const float* b2    = (const float*)d_in[14];
  const float* Wo1   = (const float*)d_in[15];
  const float* bo1   = (const float*)d_in[16];
  const float* Wo2   = (const float*)d_in[17];
  const float* bo2   = (const float*)d_in[18];
  const float* Wo3   = (const float*)d_in[19];
  const float* bo3   = (const float*)d_in[20];
  float* out = (float*)d_out;

  const int N = in_sizes[1];
  const int R = out_size / 10;
  int Ee[3] = { in_sizes[2] / 2, in_sizes[3] / 2, in_sizes[4] / 2 };
  int Emax = Ee[0] > Ee[1] ? Ee[0] : Ee[1]; if (Ee[2] > Emax) Emax = Ee[2];

  // ---- carve workspace ----
  size_t cur_off = 0;
  char* base = (char*)d_ws;
  auto carve = [&](size_t bytes) -> void* {
    cur_off = (cur_off + 255) & ~(size_t)255;
    void* p = base + cur_off; cur_off += bytes; return p;
  };
  _Float16* xh    = (_Float16*)carve((size_t)N * 32 * 2);
  _Float16* bufA  = (_Float16*)carve((size_t)N * 64 * 2);
  _Float16* bufB  = (_Float16*)carve((size_t)N * 64 * 2);
  float*    xcomb = (float*)   carve((size_t)N * 64 * 4);
  float*    al_s  = (float*)   carve((size_t)N * 4 * 4);
  float*    al_d  = (float*)   carve((size_t)N * 4 * 4);
  int*      cnt   = (int*)     carve((size_t)N * 4);
  int*      offs  = (int*)     carve((size_t)N * 4);
  int*      curs  = (int*)     carve((size_t)N * 4);
  int*      bsums = (int*)     carve((size_t)SCAN_B * 4);
  int*      list  = (int*)     carve((size_t)(Emax + N) * 4);
  _Float16* Bp    = (_Float16*)carve(4096 * 2);
  float*    ew    = (float*)   carve(64);
  float*    sums  = (float*)   carve((size_t)R * 64 * 4);
  float*    cnts  = (float*)   carve((size_t)R * 4);
  float*    hq1   = (float*)   carve((size_t)R * 64 * 4);
  float*    hq2   = (float*)   carve((size_t)R * 32 * 4);

  const int B = 256;
  const unsigned nbScan = cdiv(N, SCAN_B);

  // ---- setup ----
  k_ew<<<1, 1, 0, stream>>>(eattn, ew);
  k_build_x<<<cdiv((long)N * 32, B), B, 0, stream>>>(nf, ntyp, emb, xh, N);
  k_zero_f32<<<cdiv((long)N * 64, B), B, 0, stream>>>(xcomb, N * 64);

  // ---- per edge type ----
  for (int i = 0; i < 3; ++i){
    const int* ei = ei3[i];
    const int  E  = Ee[i];

    // CSR build (shared by both layers of this edge type)
    k_fill_i32<<<cdiv(N, B), B, 0, stream>>>(cnt, 0, N);
    k_count<<<cdiv((long)E + N, B), B, 0, stream>>>(ei, E, N, cnt);
    k_scan_blk<<<nbScan, SCAN_B, 0, stream>>>(cnt, offs, bsums, N);
    k_scan_top<<<1, SCAN_B, 0, stream>>>(bsums, (int)nbScan);
    k_scan_add<<<nbScan, SCAN_B, 0, stream>>>(offs, bsums, N);
    k_fill_i32<<<cdiv(N, B), B, 0, stream>>>(curs, 0, N);
    k_fillcsr<<<cdiv((long)E + N, B), B, 0, stream>>>(ei, E, N, offs, cnt, curs, list);

    // Layer 1: WMMA GEMM (K=32 padded) + fused GAT aggregation (H=4) + ELU
    k_pack_w<<<cdiv(2048, B), B, 0, stream>>>(W1 + (size_t)i * 23 * 64, Bp, 23);
    k_gemm<1><<<cdiv(N, 64), 128, 0, stream>>>(xh, 32, Bp, bufA, N);
    k_al4<<<cdiv((long)N * 4, B), B, 0, stream>>>(bufA, as1 + i * 64, ad1 + i * 64, al_s, al_d, N);
    k_agg1<<<cdiv((long)N * 4, B), B, 0, stream>>>(offs, cnt, list, al_s, al_d, bufA,
                                                   b1 + i * 64, bufB, N);

    // Layer 2: WMMA GEMM (K=64) + fused GAT aggregation (H=1) + ew accumulate
    k_pack_w<<<cdiv(4096, B), B, 0, stream>>>(W2 + (size_t)i * 64 * 64, Bp, 64);
    k_gemm<2><<<cdiv(N, 64), 128, 0, stream>>>(bufB, 64, Bp, bufA, N);
    k_al1<<<cdiv(N, B), B, 0, stream>>>(bufA, as2 + i * 64, ad2 + i * 64, al_s, al_d, N);
    k_agg2<<<cdiv((long)N * 4, B), B, 0, stream>>>(offs, cnt, list, al_s, al_d, bufA,
                                                   b2 + i * 64, ew, i, xcomb, N);
  }

  // ---- global Q readout ----
  k_zero_f32<<<cdiv((long)R * 64, B), B, 0, stream>>>(sums, R * 64);
  k_zero_f32<<<cdiv(R, B), B, 0, stream>>>(cnts, R);
  k_qsum<<<cdiv(Ee[0], B), B, 0, stream>>>(ei3[0], Ee[0], R, xcomb, sums, cnts);
  k_mlp1<<<cdiv((long)R * 64, B), B, 0, stream>>>(xcomb, sums, cnts, Wo1, bo1, hq1, R);
  k_mlp2<<<cdiv((long)R * 32, B), B, 0, stream>>>(hq1, Wo2, bo2, hq2, R);
  k_mlp3<<<cdiv((long)R * 10, B), B, 0, stream>>>(hq2, Wo3, bo3, out, R);
}
